// LSTMModel_76587856822916
// MI455X (gfx1250) — compile-verified, matrix operand order checked by reading, not run on hardware
//
#include <hip/hip_runtime.h>
#include <math.h>

// ---------------------------------------------------------------------------
// Model dims (match the reference)
#define Hh    512
#define EMBd  300
#define TAGEd 20
#define TREd  20
#define Bn    128
#define Sn    256
#define Tn    512
#define Rn    128          // S/2
#define G4H   2048         // 4*H
#define NTR   100          // NTRANS
// ---------------------------------------------------------------------------

typedef __bf16 bf16;
typedef __attribute__((ext_vector_type(16))) __bf16 v16bf;
typedef __attribute__((ext_vector_type(8)))  __bf16 v8bf;
typedef __attribute__((ext_vector_type(8)))  float  v8f;

__device__ __forceinline__ float sigf(float x) { return 1.0f / (1.0f + expf(-x)); }

// ---------------------------------------------------------------------------
// CDNA5 async global->LDS staging (ASYNCcnt path), with safe fallback.
// Builtin signature (from hipcc diagnostic): params are b128 = v4i pointers,
// src in the global (AS1 / __device__) space, dst in LDS (AS3).
// ---------------------------------------------------------------------------
#if __has_builtin(__builtin_amdgcn_global_load_async_to_lds_b128)
#define ASYNC_LDS 1
#else
#define ASYNC_LDS 0
#endif

typedef int vi4 __attribute__((vector_size(16)));
typedef __attribute__((address_space(1))) vi4* g_vi4_ptr;
typedef __attribute__((address_space(3))) vi4* l_vi4_ptr;

__device__ __forceinline__ void async_ld16(const void* g, void* lds_generic)
{
#if ASYNC_LDS
    // AS1 pointer: same 64-bit value as generic for global memory.
    // AS3 pointer: low 32 bits of the generic shared-aperture address
    // (flat->LDS mapping keeps the low 32 bits, ISA 10.2).
    __builtin_amdgcn_global_load_async_to_lds_b128(
        (g_vi4_ptr)(unsigned long long)(uintptr_t)g,
        (l_vi4_ptr)(unsigned int)(uintptr_t)lds_generic,
        0, 0);
#else
    (void)g; (void)lds_generic;
#endif
}

__device__ __forceinline__ void async_wait0()
{
#if ASYNC_LDS
#if __has_builtin(__builtin_amdgcn_s_wait_asynccnt)
    __builtin_amdgcn_s_wait_asynccnt(0);
#else
    asm volatile("s_wait_asynccnt 0x0" ::: "memory");
#endif
#endif
}

// ---------------------------------------------------------------------------
// Generic bf16 WMMA GEMM:
//   C[M,N](f32/bf16) = concat(Ah[M,hK], Ax[M,xK]) @ W[N, hK+xK]^T + bias[N]
// Block = 256 thr = 8 waves -> 64(M) x 128(N) tile; wave = 32x32 (2x2 wmma).
// K staged in LDS in chunks of 32 (all K are multiples of 32 by construction).
// Staging uses global_load_async_to_lds_b128 when the toolchain exposes it
// (guardN==0), else the synchronous VGPR round-trip.
// Fragment gathers follow the CDNA5 16-bit A/B VGPR layouts (ISA 7.12.2):
//   lanes 0-15 hold K=[0..7]+[16..23], lanes 16-31 hold K=[8..15]+[24..31].
// ---------------------------------------------------------------------------
#define AST 40   // LDS row stride in bf16 (80B: 16B-aligned, conflict-free)

__global__ __launch_bounds__(256)
void wmma_gemm_kernel(const bf16* __restrict__ Ah, int hK,
                      const bf16* __restrict__ Ax, int xK, long xstride,
                      const bf16* __restrict__ W,  int ldw,
                      const float* __restrict__ bias,
                      float* __restrict__ Cf, bf16* __restrict__ Cb, int ldc,
                      int N, int act, int guardN)
{
    __shared__ bf16 As[64 * AST];
    __shared__ bf16 Bs[128 * AST];

    const int  tid  = threadIdx.x;
    const int  K    = hK + xK;
    const long m0   = (long)blockIdx.y * 64;
    const int  n0   = blockIdx.x * 128;

    const int  wv   = tid >> 5;
    const int  lane = tid & 31;
    const int  wm   = (wv >> 2) * 32;     // wave M offset within block tile
    const int  wn   = (wv & 3) * 32;      // wave N offset
    const int  hv   = lane >> 4;          // lane half
    const int  rr   = lane & 15;
    const int  kb   = hv * 8;             // K base for this half

#if ASYNC_LDS
    const bool sync_path = (guardN != 0);
#else
    const bool sync_path = true;
#endif

    v8f acc[2][2];
#pragma unroll
    for (int i = 0; i < 2; ++i)
#pragma unroll
        for (int j = 0; j < 2; ++j)
#pragma unroll
            for (int e = 0; e < 8; ++e) acc[i][j][e] = 0.0f;

    const int arow = tid >> 2;            // 0..63
    const int acol = (tid & 3) * 8;       // 0,8,16,24

    for (int kc = 0; kc < K; kc += 32) {
        const long gm = m0 + arow;
        const int  k  = kc + acol;
        const bf16* asrc = (k < hK) ? (Ah + gm * (long)hK + k)
                                    : (Ax + gm * xstride + (k - hK));

        // prefetch next K-chunk of the (L2-resident) weight tile
        if (kc + 32 < K)
            __builtin_prefetch((const void*)(W + (long)(n0 + arow) * ldw + k + 32), 0, 1);

        if (!sync_path) {
            // ---- async ASYNCcnt-tracked staging, no VGPR round-trip ----
            async_ld16(asrc, As + arow * AST + acol);
#pragma unroll
            for (int i = 0; i < 2; ++i) {
                const int brow = arow + i * 64;
                async_ld16(W + (long)(n0 + brow) * ldw + k, Bs + brow * AST + acol);
            }
            async_wait0();
        } else {
            // ---- synchronous staging (fallback / ragged-N tiles) ----
            *(v8bf*)(As + arow * AST + acol) = *(const v8bf*)asrc;
#pragma unroll
            for (int i = 0; i < 2; ++i) {
                const int brow = arow + i * 64;
                const int gn   = n0 + brow;
                v8bf v;
                if (gn < N) {
                    v = *(const v8bf*)(W + (long)gn * ldw + k);
                } else {
#pragma unroll
                    for (int e = 0; e < 8; ++e) v[e] = (bf16)0.0f;
                }
                *(v8bf*)(Bs + brow * AST + acol) = v;
            }
        }
        __syncthreads();

        // ---- gather fragments (ds_load_b128 pairs) and WMMA ----
        v16bf af[2], bg[2];
#pragma unroll
        for (int i = 0; i < 2; ++i) {
            const int r  = wm + i * 16 + rr;
            v8bf lo = *(const v8bf*)(As + r * AST + kb);
            v8bf hi = *(const v8bf*)(As + r * AST + kb + 16);
#pragma unroll
            for (int e = 0; e < 8; ++e) { af[i][e] = lo[e]; af[i][8 + e] = hi[e]; }
        }
#pragma unroll
        for (int j = 0; j < 2; ++j) {
            const int c  = wn + j * 16 + rr;
            v8bf lo = *(const v8bf*)(Bs + c * AST + kb);
            v8bf hi = *(const v8bf*)(Bs + c * AST + kb + 16);
#pragma unroll
            for (int e = 0; e < 8; ++e) { bg[j][e] = lo[e]; bg[j][8 + e] = hi[e]; }
        }
#pragma unroll
        for (int i = 0; i < 2; ++i)
#pragma unroll
            for (int j = 0; j < 2; ++j)
                acc[i][j] = __builtin_amdgcn_wmma_f32_16x16x32_bf16(
                    false, af[i], false, bg[j], (short)0, acc[i][j], false, false);
        __syncthreads();
    }

    // ---- epilogue: D layout (VGPR e -> M = hv*8+e, N = rr) ----
#pragma unroll
    for (int i = 0; i < 2; ++i)
#pragma unroll
        for (int j = 0; j < 2; ++j) {
            const int col = n0 + wn + j * 16 + rr;
            if (col < N) {
                const float bval = bias ? bias[col] : 0.0f;
#pragma unroll
                for (int e = 0; e < 8; ++e) {
                    const long row = m0 + wm + i * 16 + hv * 8 + e;
                    float v = acc[i][j][e] + bval;
                    if (act == 1) v = tanhf(v);
                    if (Cf) Cf[row * (long)ldc + col] = v;
                    if (Cb) Cb[row * (long)ldc + col] = (bf16)v;
                }
            }
        }
}

// ---------------------------------------------------------------------------
// LSTM gate elementwise: g[M,2048] (i,f,g,o) -> c, h(bf16), optional archive
// ---------------------------------------------------------------------------
__global__ void lstm_elem_kernel(const float* __restrict__ g,
                                 float* __restrict__ c,
                                 bf16* __restrict__ hb,
                                 bf16* __restrict__ arch, long archStride, long M)
{
    const long idx = (long)blockIdx.x * 256 + threadIdx.x;
    if (idx >= M * Hh) return;
    const long m = idx >> 9;
    const int  j = (int)(idx & (Hh - 1));
    const float* gr = g + m * (long)G4H;
    const float gi = gr[j], gf = gr[Hh + j], gg = gr[2 * Hh + j], go = gr[3 * Hh + j];
    const float cv = sigf(gf) * c[idx] + sigf(gi) * tanhf(gg);
    c[idx] = cv;
    const float h = sigf(go) * tanhf(cv);
    hb[idx] = (bf16)h;
    if (arch) arch[m * archStride + j] = (bf16)h;
}

// ---------------------------------------------------------------------------
// Weight pack: out[n, 0:hK]=Whh, out[n, hK:hK+Din]=Wih, rest zero (bf16)
// ---------------------------------------------------------------------------
__global__ void pack_w_kernel(const float* __restrict__ Whh, int hK,
                              const float* __restrict__ Wih, int Din,
                              bf16* __restrict__ out, int ldw, long Nrows)
{
    const long idx = (long)blockIdx.x * 256 + threadIdx.x;
    if (idx >= Nrows * ldw) return;
    const long n = idx / ldw;
    const int  k = (int)(idx % ldw);
    float v = 0.0f;
    if (k < hK)                 v = Whh[n * (long)hK + k];
    else if (k - hK < Din)      v = Wih[n * (long)Din + (k - hK)];
    out[idx] = (bf16)v;
}

// word input: concat(emb[word_idx], tag_emb[tag_idx]) -> bf16 [B*S, 320]
__global__ void build_word_in_kernel(const int* __restrict__ widx,
                                     const int* __restrict__ tidx,
                                     const float* __restrict__ emb,
                                     const float* __restrict__ tag_emb,
                                     bf16* __restrict__ out)
{
    const long idx = (long)blockIdx.x * 256 + threadIdx.x;
    const int D = EMBd + TAGEd;
    if (idx >= (long)Bn * Sn * D) return;
    const long n = idx / D;
    const int  j = (int)(idx % D);
    float v = (j < EMBd) ? emb[(long)widx[n] * EMBd + j]
                         : tag_emb[(long)tidx[n] * TAGEd + (j - EMBd)];
    out[idx] = (bf16)v;
}

// transition input: tr_emb gather, K padded 20->32 with zeros
__global__ void build_tr_in_kernel(const int* __restrict__ idxs,
                                   const float* __restrict__ tr_emb,
                                   bf16* __restrict__ out)
{
    const long idx = (long)blockIdx.x * 256 + threadIdx.x;
    if (idx >= (long)Bn * Tn * 32) return;
    const long n = idx >> 5;
    const int  j = (int)(idx & 31);
    float v = (j < TREd) ? tr_emb[(long)idxs[n] * TREd + j] : 0.0f;
    out[idx] = (bf16)v;
}

// label input: const_emb gather -> bf16 [B*R, 512]
__global__ void build_lab_kernel(const int* __restrict__ lidx,
                                 const float* __restrict__ const_emb,
                                 bf16* __restrict__ out)
{
    const long idx = (long)blockIdx.x * 256 + threadIdx.x;
    if (idx >= (long)Bn * Rn * Hh) return;
    const long n = idx >> 9;
    const int  j = (int)(idx & (Hh - 1));
    out[idx] = (bf16)const_emb[(long)lidx[n] * Hh + j];
}

// children gather: ch[b*R + r] = c0[b*S + 2r + sel]
__global__ void build_ch_kernel(const bf16* __restrict__ c0b,
                                bf16* __restrict__ out, int sel)
{
    const long idx = (long)blockIdx.x * 256 + threadIdx.x;
    if (idx >= (long)Bn * Rn * Hh) return;
    const long n = idx >> 9;
    const int  j = (int)(idx & (Hh - 1));
    const long b = n >> 7, r = n & (Rn - 1);
    out[idx] = c0b[((b * Sn + 2 * r + sel) << 9) + j];
}

// bf16 row-block copy (builds hx = [word_h_last | tr_h | const_h])
__global__ void copy_rows_kernel(bf16* __restrict__ dst, long ldd,
                                 const bf16* __restrict__ src, long lds,
                                 long rows, int cols)
{
    const long idx = (long)blockIdx.x * 256 + threadIdx.x;
    if (idx >= rows * cols) return;
    const long r = idx / cols;
    const int  c = (int)(idx % cols);
    dst[r * ldd + c] = src[r * lds + c];
}

// ---------------------------------------------------------------------------
extern "C" void kernel_launch(void* const* d_in, const int* in_sizes, int n_in,
                              void* d_out, int out_size, void* d_ws, size_t ws_size,
                              hipStream_t stream)
{
    const int*   word_idx   = (const int*)d_in[0];
    const int*   tag_idx    = (const int*)d_in[1];
    const int*   trans_idx  = (const int*)d_in[2];
    const int*   label_idx  = (const int*)d_in[3];
    const float* emb        = (const float*)d_in[4];
    const float* tag_emb    = (const float*)d_in[5];
    const float* tr_emb     = (const float*)d_in[6];
    const float* const_emb  = (const float*)d_in[7];
    const float* word_Wih = (const float*)d_in[8],  *word_Whh = (const float*)d_in[9],  *word_b = (const float*)d_in[10];
    const float* tr_Wih   = (const float*)d_in[11], *tr_Whh   = (const float*)d_in[12], *tr_b   = (const float*)d_in[13];
    const float* con_Wih  = (const float*)d_in[14], *con_Whh  = (const float*)d_in[15], *con_b  = (const float*)d_in[16];
    const float* fwd_Wih  = (const float*)d_in[17], *fwd_Whh  = (const float*)d_in[18], *fwd_b  = (const float*)d_in[19];
    const float* bwd_Wih  = (const float*)d_in[20], *bwd_Whh  = (const float*)d_in[21], *bwd_b  = (const float*)d_in[22];
    const float* w2c_W = (const float*)d_in[23], *w2c_b = (const float*)d_in[24];
    const float* red_W = (const float*)d_in[25], *red_b = (const float*)d_in[26];
    const float* out_W = (const float*)d_in[27], *out_b = (const float*)d_in[28];
    float* dOut = (float*)d_out;

    // --- carve workspace ---
    size_t off = 0;
    auto carve = [&](size_t bytes) -> void* {
        off = (off + 255) & ~(size_t)255;
        void* p = (char*)d_ws + off;
        off += bytes;
        return p;
    };
    bf16* wordW  = (bf16*)carve((size_t)G4H * 832 * 2);
    bf16* trW    = (bf16*)carve((size_t)G4H * 544 * 2);
    bf16* conW   = (bf16*)carve((size_t)G4H * 1024 * 2);
    bf16* fwdW   = (bf16*)carve((size_t)G4H * 1024 * 2);
    bf16* bwdW   = (bf16*)carve((size_t)G4H * 1024 * 2);
    bf16* w2cW   = (bf16*)carve((size_t)Hh * Hh * 2);
    bf16* redW   = (bf16*)carve((size_t)Hh * 1024 * 2);
    bf16* outWb  = (bf16*)carve((size_t)NTR * 1536 * 2);

    bf16* word_in   = (bf16*)carve((size_t)Bn * Sn * 320 * 2);
    bf16* tr_in     = (bf16*)carve((size_t)Bn * Tn * 32 * 2);
    bf16* word_hseq = (bf16*)carve((size_t)Bn * Sn * Hh * 2);
    bf16* c0b       = (bf16*)carve((size_t)Bn * Sn * Hh * 2);
    bf16* lab       = (bf16*)carve((size_t)Bn * Rn * Hh * 2);
    bf16* ch0       = (bf16*)carve((size_t)Bn * Rn * Hh * 2);
    bf16* ch1       = (bf16*)carve((size_t)Bn * Rn * Hh * 2);
    bf16* redin     = (bf16*)carve((size_t)Bn * Rn * 1024 * 2);
    bf16* redout    = (bf16*)carve((size_t)Bn * Rn * Hh * 2);
    bf16* hx        = (bf16*)carve((size_t)Bn * 1536 * 2);

    float* gsmall  = (float*)carve((size_t)Bn * G4H * 4);
    float* gbig    = (float*)carve((size_t)Bn * Rn * G4H * 4);
    bf16*  word_hb = (bf16*)carve((size_t)Bn * Hh * 2);
    bf16*  tr_hb   = (bf16*)carve((size_t)Bn * Hh * 2);
    bf16*  con_hb  = (bf16*)carve((size_t)Bn * Hh * 2);
    float* word_c  = (float*)carve((size_t)Bn * Hh * 4);
    float* tr_c    = (float*)carve((size_t)Bn * Hh * 4);
    float* con_c   = (float*)carve((size_t)Bn * Hh * 4);
    bf16*  fb_hb   = (bf16*)carve((size_t)Bn * Rn * Hh * 2);
    float* fb_c    = (float*)carve((size_t)Bn * Rn * Hh * 4);

    auto gemm = [&](const bf16* Ah, int hK, const bf16* Ax, int xK, long xstride,
                    const bf16* W, int ldw, const float* bias,
                    float* Cf, bf16* Cb, int ldc, long M, int N, int act) {
        dim3 grid((N + 127) / 128, (unsigned)(M / 64));
        const int guardN = (N & 127) ? 1 : 0;
        wmma_gemm_kernel<<<grid, 256, 0, stream>>>(Ah, hK, Ax, xK, xstride,
                                                   W, ldw, bias, Cf, Cb, ldc, N, act, guardN);
    };
    auto elem = [&](const float* g, float* c, bf16* hb, bf16* arch, long archStride, long M) {
        long tot = M * Hh;
        lstm_elem_kernel<<<(unsigned)((tot + 255) / 256), 256, 0, stream>>>(g, c, hb, arch, archStride, M);
    };
    auto ew = [&](long tot) { return (unsigned)((tot + 255) / 256); };

    // --- weight packing (f32 -> bf16, concat [Whh|Wih], K padded to x32) ---
    pack_w_kernel<<<ew((long)G4H * 832), 256, 0, stream>>>(word_Whh, Hh, word_Wih, 320, wordW, 832, G4H);
    pack_w_kernel<<<ew((long)G4H * 544), 256, 0, stream>>>(tr_Whh, Hh, tr_Wih, TREd, trW, 544, G4H);
    pack_w_kernel<<<ew((long)G4H * 1024), 256, 0, stream>>>(con_Whh, Hh, con_Wih, Hh, conW, 1024, G4H);
    pack_w_kernel<<<ew((long)G4H * 1024), 256, 0, stream>>>(fwd_Whh, Hh, fwd_Wih, Hh, fwdW, 1024, G4H);
    pack_w_kernel<<<ew((long)G4H * 1024), 256, 0, stream>>>(bwd_Whh, Hh, bwd_Wih, Hh, bwdW, 1024, G4H);
    pack_w_kernel<<<ew((long)Hh * Hh), 256, 0, stream>>>(nullptr, 0, w2c_W, Hh, w2cW, Hh, Hh);
    pack_w_kernel<<<ew((long)Hh * 1024), 256, 0, stream>>>(nullptr, 0, red_W, 1024, redW, 1024, Hh);
    pack_w_kernel<<<ew((long)NTR * 1536), 256, 0, stream>>>(nullptr, 0, out_W, 1536, outWb, 1536, NTR);

    // --- input gathers ---
    build_word_in_kernel<<<ew((long)Bn * Sn * 320), 256, 0, stream>>>(word_idx, tag_idx, emb, tag_emb, word_in);
    build_tr_in_kernel<<<ew((long)Bn * Tn * 32), 256, 0, stream>>>(trans_idx, tr_emb, tr_in);
    build_lab_kernel<<<ew((long)Bn * Rn * Hh), 256, 0, stream>>>(label_idx, const_emb, lab);

    // --- zero-init LSTM states (every call: deterministic) ---
    (void)hipMemsetAsync(word_c, 0, (size_t)Bn * Hh * 4, stream);
    (void)hipMemsetAsync(word_hb, 0, (size_t)Bn * Hh * 2, stream);
    (void)hipMemsetAsync(tr_c, 0, (size_t)Bn * Hh * 4, stream);
    (void)hipMemsetAsync(tr_hb, 0, (size_t)Bn * Hh * 2, stream);
    (void)hipMemsetAsync(con_c, 0, (size_t)Bn * Hh * 4, stream);
    (void)hipMemsetAsync(con_hb, 0, (size_t)Bn * Hh * 2, stream);

    // --- word LSTM: 256 steps, g = [h|x_t] @ [Whh|Wih]^T + b ---
    for (int t = 0; t < Sn; ++t) {
        gemm(word_hb, Hh, word_in + (long)t * 320, 320, (long)Sn * 320,
             wordW, 832, word_b, gsmall, nullptr, G4H, Bn, G4H, 0);
        elem(gsmall, word_c, word_hb, word_hseq + (long)t * Hh, (long)Sn * Hh, Bn);
    }

    // --- transition LSTM: 512 steps ---
    for (int t = 0; t < Tn; ++t) {
        gemm(tr_hb, Hh, tr_in + (long)t * 32, 32, (long)Tn * 32,
             trW, 544, tr_b, gsmall, nullptr, G4H, Bn, G4H, 0);
        elem(gsmall, tr_c, tr_hb, nullptr, 0, Bn);
    }

    // --- w2c: c0 = word_h @ w2c_W^T + b (bf16 out) ---
    gemm(nullptr, 0, word_hseq, Hh, Hh, w2cW, Hh, w2c_b,
         nullptr, c0b, Hh, (long)Bn * Sn, Hh, 0);

    build_ch_kernel<<<ew((long)Bn * Rn * Hh), 256, 0, stream>>>(c0b, ch0, 0);
    build_ch_kernel<<<ew((long)Bn * Rn * Hh), 256, 0, stream>>>(c0b, ch1, 1);

    // --- fwd reduce LSTM: 3 steps over [lab, ch0, ch1], M = 16384 ---
    {
        (void)hipMemsetAsync(fb_c, 0, (size_t)Bn * Rn * Hh * 4, stream);
        (void)hipMemsetAsync(fb_hb, 0, (size_t)Bn * Rn * Hh * 2, stream);
        const bf16* xs[3] = { lab, ch0, ch1 };
        for (int t = 0; t < 3; ++t) {
            gemm(fb_hb, Hh, xs[t], Hh, Hh, fwdW, 1024, fwd_b,
                 gbig, nullptr, G4H, (long)Bn * Rn, G4H, 0);
            elem(gbig, fb_c, fb_hb, (t == 2) ? redin : nullptr, 1024, (long)Bn * Rn);
        }
    }
    // --- bwd reduce LSTM: 3 steps over [lab, ch1, ch0] ---
    {
        (void)hipMemsetAsync(fb_c, 0, (size_t)Bn * Rn * Hh * 4, stream);
        (void)hipMemsetAsync(fb_hb, 0, (size_t)Bn * Rn * Hh * 2, stream);
        const bf16* xs[3] = { lab, ch1, ch0 };
        for (int t = 0; t < 3; ++t) {
            gemm(fb_hb, Hh, xs[t], Hh, Hh, bwdW, 1024, bwd_b,
                 gbig, nullptr, G4H, (long)Bn * Rn, G4H, 0);
            elem(gbig, fb_c, fb_hb, (t == 2) ? (redin + Hh) : nullptr, 1024, (long)Bn * Rn);
        }
    }

    // --- red = tanh([fh|bh] @ red_W^T + b) (bf16 out) ---
    gemm(nullptr, 0, redin, 1024, 1024, redW, 1024, red_b,
         nullptr, redout, Hh, (long)Bn * Rn, Hh, 1);

    // --- constituent LSTM: 128 steps over red ---
    for (int t = 0; t < Rn; ++t) {
        gemm(con_hb, Hh, redout + (long)t * Hh, Hh, (long)Rn * Hh,
             conW, 1024, con_b, gsmall, nullptr, G4H, Bn, G4H, 0);
        elem(gsmall, con_c, con_hb, nullptr, 0, Bn);
    }

    // --- hx = [word_h[:, -1] | tr_h | const_h] ---
    copy_rows_kernel<<<ew((long)Bn * Hh), 256, 0, stream>>>(hx, 1536, word_hseq + (long)(Sn - 1) * Hh, (long)Sn * Hh, Bn, Hh);
    copy_rows_kernel<<<ew((long)Bn * Hh), 256, 0, stream>>>(hx + Hh, 1536, tr_hb, Hh, Bn, Hh);
    copy_rows_kernel<<<ew((long)Bn * Hh), 256, 0, stream>>>(hx + 2 * Hh, 1536, con_hb, Hh, Bn, Hh);

    // --- out = hx @ out_W^T + out_b (f32, [128, 100]) ---
    gemm(nullptr, 0, hx, 1536, 1536, outWb, 1536, out_b,
         dOut, nullptr, NTR, Bn, NTR, 0);
}
